// LocalEquivariantLayer_5368709120838
// MI455X (gfx1250) — matrix operand order repaired and checked by reading, not compiled, not run on hardware
//
#include <hip/hip_runtime.h>

typedef _Float16 h8   __attribute__((ext_vector_type(8)));
typedef _Float16 v16h __attribute__((ext_vector_type(16)));
typedef float    v8f  __attribute__((ext_vector_type(8)));

#define NN 1024
#define HH 64

__device__ __forceinline__ float silu_f(float x) {
    float t = __builtin_amdgcn_exp2f(-1.44269504f * x);
    return x * __builtin_amdgcn_rcpf(1.0f + t);
}

// ---------------------------------------------------------------------------
// Kernel 0: per-row precompute.  A = feat@wa, B = feat@wb + msg_b1 (f16),
// plus zero the accumulation workspaces.
// ---------------------------------------------------------------------------
__global__ void prep_rows(const float* __restrict__ feat,
                          const float* __restrict__ msg_w1,
                          const float* __restrict__ msg_b1,
                          _Float16* __restrict__ apre,
                          _Float16* __restrict__ bvh,
                          float* __restrict__ aggws,
                          float* __restrict__ cupdws) {
    int r = blockIdx.x, t = threadIdx.x;
    float a = 0.f, b = 0.f;
    for (int k = 0; k < HH; ++k) {
        float f = feat[r * HH + k];
        a += f * msg_w1[k * HH + t];
        b += f * msg_w1[(HH + k) * HH + t];
    }
    b += msg_b1[t];
    apre[r * HH + t] = (_Float16)a;
    bvh [r * HH + t] = (_Float16)b;
    aggws[r * HH + t] = 0.f;
    if (t < 3) cupdws[r * 3 + t] = 0.f;
}

// ---------------------------------------------------------------------------
// Kernel 1: small constants: wc (msg_w1 row 128), cb1' = coord_b1 + msg_b2@coord_w1,
// coord_w2 -> f16.
// ---------------------------------------------------------------------------
__global__ void prep_const(const float* __restrict__ msg_w1,
                           const float* __restrict__ msg_b2,
                           const float* __restrict__ coord_w1,
                           const float* __restrict__ coord_b1,
                           const float* __restrict__ coord_w2,
                           _Float16* __restrict__ wch,
                           _Float16* __restrict__ cb1p,
                           _Float16* __restrict__ cw2h) {
    int t = threadIdx.x;
    wch[t] = (_Float16)msg_w1[2 * HH * HH + t];   // row 128 of msg_w1
    float c = coord_b1[t];
    for (int n = 0; n < HH; ++n) c += msg_b2[n] * coord_w1[n * HH + t];
    cb1p[t] = (_Float16)c;
    cw2h[t] = (_Float16)coord_w2[t];
}

// ---------------------------------------------------------------------------
// Main fused pair kernel.  One wave per block (wave32).
// Block (x,y): i-tile = 16*x .. 16*x+15, j-range = 32*y .. 32*y+31.
// ---------------------------------------------------------------------------
__global__ __launch_bounds__(32) void egnn_pairs(
        const float* __restrict__ coords,
        const float* __restrict__ msg_w2,
        const float* __restrict__ coord_w1,
        const float* __restrict__ coord_b2,
        const _Float16* __restrict__ apre,
        const _Float16* __restrict__ bvh,
        const _Float16* __restrict__ wch,
        const _Float16* __restrict__ cb1h,
        const _Float16* __restrict__ cw2h,
        float* __restrict__ aggws,
        float* __restrict__ cupdws) {
    const int lane = threadIdx.x;
    const int h = lane >> 4;          // lane half: selects K-offset +16 in B layout
    const int p = lane & 15;          // pair column within tile
    const int i0 = blockIdx.x * 16;
    const int i_me = i0 + p;

    // ---- persistent per-lane data (live across whole j loop) ----
    h8 apk[4], wc8[4], cb1r[4], cw2r[4];
#pragma unroll
    for (int q = 0; q < 4; ++q) {
        apk[q] = *(const h8*)(apre + i_me * HH + 32 * h + 8 * q);
        wc8[q] = *(const h8*)(wch + 32 * h + 8 * q);
    }
#pragma unroll
    for (int t = 0; t < 4; ++t) {     // feature n2 = 16t + 8h + r  (r = elem)
        cb1r[t] = *(const h8*)(cb1h + 16 * t + 8 * h);
        cw2r[t] = *(const h8*)(cw2h + 16 * t + 8 * h);
    }
    float cix = coords[3 * i_me], ciy = coords[3 * i_me + 1], ciz = coords[3 * i_me + 2];
    float cb2v = coord_b2[0];

    // ---- static A-fragments: W2^T and CW1^T with permuted K assignments ----
    v16h w2f[4][2], c1f[4][2];
#pragma unroll
    for (int a = 0; a < 4; ++a)
#pragma unroll
        for (int u = 0; u < 2; ++u) {
            v16h f, g;
#pragma unroll
            for (int e = 0; e < 16; ++e) {
                int pr = e >> 1, lo = e & 1;
                int kp = 16 * h + 2 * pr + lo;
                // GEMM1: logical k = 32*(kp>>4) + 16u + (kp&15)  (matches per-lane
                //        contiguous runs of the activation vector)
                int klog  = 32 * (kp >> 4) + 16 * u + (kp & 15);
                // GEMM2: logical feature = 16*(2u + ((kp&15)>>3)) + 8*(kp>>4) + (kp&7)
                //        chosen so GEMM1's D layout == GEMM2's B layout (no shuffles)
                int kfeat = 16 * (2 * u + ((kp & 15) >> 3)) + 8 * (kp >> 4) + (kp & 7);
                f[e] = (_Float16)msg_w2  [klog  * HH + 16 * a + p];
                g[e] = (_Float16)coord_w1[kfeat * HH + 16 * a + p];
            }
            w2f[a][u] = f;
            c1f[a][u] = g;
        }

    const v8f zero = {};
    v8f agg[4];
#pragma unroll
    for (int a = 0; a < 4; ++a) agg[a] = zero;
    float cux = 0.f, cuy = 0.f, cuz = 0.f;

    const int jbase = blockIdx.y * 32;
    for (int jj = 0; jj < 32; ++jj) {
        int j = jbase + jj;

        h8 bq[4];
#pragma unroll
        for (int q = 0; q < 4; ++q)
            bq[q] = *(const h8*)(bvh + j * HH + 32 * h + 8 * q);

        float cjx = coords[3 * j], cjy = coords[3 * j + 1], cjz = coords[3 * j + 2];
        float rx = cix - cjx, ry = ciy - cjy, rz = ciz - cjz;
        float r2 = rx * rx + ry * ry + rz * rz;
        float dist = (r2 > 0.f) ? __builtin_amdgcn_sqrtf(r2) : 0.f;
        _Float16 dh = (_Float16)dist;
        h8 dsp = {dh, dh, dh, dh, dh, dh, dh, dh};

        // pre = A[i] + B[j] + dist*wc  (packed f16), then silu -> B1 fragments
        v16h b1[2];
#pragma unroll
        for (int q = 0; q < 4; ++q) {
            h8 pre = apk[q] + bq[q] + wc8[q] * dsp;
#pragma unroll
            for (int c = 0; c < 8; ++c) {
                float xv = (float)pre[c];
                b1[q >> 1][8 * (q & 1) + c] = (_Float16)silu_f(xv);
            }
        }

        // GEMM 1: msgs^T (rows = feature n, cols = pair p)
        v8f dd[4];
#pragma unroll
        for (int a = 0; a < 4; ++a) {
            dd[a] = __builtin_amdgcn_wmma_f32_16x16x32_f16(false, w2f[a][0], false, b1[0],
                                                           (short)0, zero, false, false);
            dd[a] = __builtin_amdgcn_wmma_f32_16x16x32_f16(false, w2f[a][1], false, b1[1],
                                                           (short)0, dd[a], false, false);
        }

        // agg accumulation with diagonal mask (column i_me == j zeroed)
        float mult = (i_me == j) ? 0.f : 1.f;
#pragma unroll
        for (int a = 0; a < 4; ++a)
#pragma unroll
            for (int r = 0; r < 8; ++r) agg[a][r] += dd[a][r] * mult;

        // D -> B2: pure in-lane repack (feature permutation makes layouts match)
        v16h b2[2];
#pragma unroll
        for (int u = 0; u < 2; ++u)
#pragma unroll
            for (int s = 0; s < 8; ++s) {
                int t = 2 * u + (s >> 2), rr = 2 * (s & 3);
                b2[u][2 * s]     = (_Float16)dd[t][rr];
                b2[u][2 * s + 1] = (_Float16)dd[t][rr + 1];
            }

        // GEMM 2: y^T = CW1^T x msgs^T
        v8f yy[4];
#pragma unroll
        for (int a = 0; a < 4; ++a) {
            yy[a] = __builtin_amdgcn_wmma_f32_16x16x32_f16(false, c1f[a][0], false, b2[0],
                                                           (short)0, zero, false, false);
            yy[a] = __builtin_amdgcn_wmma_f32_16x16x32_f16(false, c1f[a][1], false, b2[1],
                                                           (short)0, yy[a], false, false);
        }

        // scalar[p] = silu(y + cb1') . cw2 + cb2  (half-wave partial + one xor-16)
        float part = 0.f;
#pragma unroll
        for (int t = 0; t < 4; ++t)
#pragma unroll
            for (int r = 0; r < 8; ++r) {
                float yv = yy[t][r] + (float)cb1r[t][r];
                part += silu_f(yv) * (float)cw2r[t][r];
            }
        float scal = part + __shfl_xor(part, 16, 32) + cb2v;

        cux += scal * rx;
        cuy += scal * ry;
        cuz += scal * rz;
    }

    // merge: feature index of agg[t][r] at this lane-half is n = 16t + 8h + r
#pragma unroll
    for (int t = 0; t < 4; ++t)
#pragma unroll
        for (int r = 0; r < 8; ++r)
            atomicAdd(&aggws[i_me * HH + 16 * t + 8 * h + r], agg[t][r]);
    if (lane < 16) {
        atomicAdd(&cupdws[i_me * 3 + 0], cux);
        atomicAdd(&cupdws[i_me * 3 + 1], cuy);
        atomicAdd(&cupdws[i_me * 3 + 2], cuz);
    }
}

// ---------------------------------------------------------------------------
// Finalize: node MLP (adds back (N-1)*msg_b2 folded out of the loop) + coords.
// ---------------------------------------------------------------------------
__global__ void finalize(const float* __restrict__ feat,
                         const float* __restrict__ coords,
                         const float* __restrict__ t_emb,
                         const float* __restrict__ msg_b2,
                         const float* __restrict__ feat_w1,
                         const float* __restrict__ feat_b1,
                         const float* __restrict__ feat_w2,
                         const float* __restrict__ feat_b2,
                         const float* __restrict__ aggws,
                         const float* __restrict__ cupdws,
                         float* __restrict__ out) {
    __shared__ float fin[2 * HH];
    __shared__ float hid[HH];
    int i = blockIdx.x, t = threadIdx.x;
    fin[t]      = feat[i * HH + t] + aggws[i * HH + t] + (float)(NN - 1) * msg_b2[t];
    fin[HH + t] = t_emb[t];
    __syncthreads();
    float s = feat_b1[t];
    for (int k = 0; k < 2 * HH; ++k) s += fin[k] * feat_w1[k * HH + t];
    hid[t] = silu_f(s);
    __syncthreads();
    float o = feat_b2[t];
    for (int k = 0; k < HH; ++k) o += hid[k] * feat_w2[k * HH + t];
    out[i * HH + t] = o;
    if (t < 3) out[NN * HH + i * 3 + t] = coords[i * 3 + t] + cupdws[i * 3 + t];
}

// ---------------------------------------------------------------------------
extern "C" void kernel_launch(void* const* d_in, const int* in_sizes, int n_in,
                              void* d_out, int out_size, void* d_ws, size_t ws_size,
                              hipStream_t stream) {
    const float* feat     = (const float*)d_in[0];
    const float* coords   = (const float*)d_in[1];
    const float* t_emb    = (const float*)d_in[2];
    const float* msg_w1   = (const float*)d_in[3];
    const float* msg_b1   = (const float*)d_in[4];
    const float* msg_w2   = (const float*)d_in[5];
    const float* msg_b2   = (const float*)d_in[6];
    const float* feat_w1  = (const float*)d_in[7];
    const float* feat_b1  = (const float*)d_in[8];
    const float* feat_w2  = (const float*)d_in[9];
    const float* feat_b2  = (const float*)d_in[10];
    const float* coord_w1 = (const float*)d_in[11];
    const float* coord_b1 = (const float*)d_in[12];
    const float* coord_w2 = (const float*)d_in[13];
    const float* coord_b2 = (const float*)d_in[14];

    char* ws = (char*)d_ws;
    float*     aggws  = (float*)(ws);                    // 1024*64*4   = 262144
    float*     cupdws = (float*)(ws + 262144);           // 1024*3*4    =  12288
    _Float16*  apre   = (_Float16*)(ws + 274432);        // 1024*64*2   = 131072
    _Float16*  bvh    = (_Float16*)(ws + 405504);        // 1024*64*2   = 131072
    _Float16*  cb1p   = (_Float16*)(ws + 536576);        // 64*2
    _Float16*  wch    = (_Float16*)(ws + 536704);        // 64*2
    _Float16*  cw2h   = (_Float16*)(ws + 536832);        // 64*2
    float* out = (float*)d_out;

    prep_rows <<<dim3(NN), dim3(HH), 0, stream>>>(feat, msg_w1, msg_b1, apre, bvh, aggws, cupdws);
    prep_const<<<dim3(1),  dim3(HH), 0, stream>>>(msg_w1, msg_b2, coord_w1, coord_b1, coord_w2,
                                                  wch, cb1p, cw2h);
    egnn_pairs<<<dim3(NN / 16, 32), dim3(32), 0, stream>>>(coords, msg_w2, coord_w1, coord_b2,
                                                           apre, bvh, wch, cb1p, cw2h,
                                                           aggws, cupdws);
    finalize  <<<dim3(NN), dim3(HH), 0, stream>>>(feat, coords, t_emb, msg_b2,
                                                  feat_w1, feat_b1, feat_w2, feat_b2,
                                                  aggws, cupdws, out);
}
